// CausalSelfAttention_18442589569148
// MI455X (gfx1250) — matrix operand order repaired and checked by reading, not compiled, not run on hardware
//
#include <hip/hip_runtime.h>
#include <hip/hip_bf16.h>

// ---------------------------------------------------------------------------
// Causal self-attention forward for MI455X (gfx1250, wave32, WMMA).
// All matmuls use v_wmma_f32_16x16x32_bf16 (bf16 in, fp32 accumulate),
// softmax in fp32 with flash-style online renormalization.
// GEMMs use 64x64 wave tiles (16 WMMA / 8 fragment loads per k-step,
// ~32 FLOP/B from L2) to stay off the L2 bandwidth ceiling.
// ---------------------------------------------------------------------------

typedef __bf16 bf16;
typedef __attribute__((ext_vector_type(16))) __bf16 v16bf;
typedef __attribute__((ext_vector_type(8)))  __bf16 v8bf;
typedef __attribute__((ext_vector_type(8)))  float  v8f;

#define B_  2
#define T_  2048
#define C_  1024
#define H_  16
#define DH_ 64
#define M_  (B_*T_)      // 4096 rows (flattened B*T)
#define N1_ (3*C_)       // 3072 qkv output features

__device__ __forceinline__ unsigned short f2bf_u(float f) {
  unsigned u = __builtin_bit_cast(unsigned, f);
  u += 0x7FFFu + ((u >> 16) & 1u);          // round-to-nearest-even
  return (unsigned short)(u >> 16);
}
__device__ __forceinline__ bf16 f2bf(float f) {
  unsigned short s = f2bf_u(f);
  return __builtin_bit_cast(bf16, s);
}
// wave32 lane broadcast via ds_bpermute (index is bytes, bits [6:2] used)
__device__ __forceinline__ float lane_bcast(float v, int srcLane) {
  int r = __builtin_amdgcn_ds_bpermute(srcLane << 2, __builtin_bit_cast(int, v));
  return __builtin_bit_cast(float, r);
}

// A fragment (16x32 bf16, ISA 7.12.2): lane L holds row m (passed in as `row`),
// K-halves split by lane>>4: element s = k {16h + s} (contiguous, matches B).
// One contiguous 32B run -> two b128 loads.
__device__ __forceinline__ v16bf load_a_frag(const bf16* p, int ld, int row, int kk, int lane) {
  int hh = lane >> 4;
  const bf16* base = p + (long)row * ld + kk + 16 * hh;
  v8bf lo = *(const v8bf*)(base);
  v8bf hi = *(const v8bf*)(base + 8);
  return __builtin_shufflevector(lo, hi, 0,1,2,3,4,5,6,7,8,9,10,11,12,13,14,15);
}
// B fragment (32x16 bf16): lane L supplies column n = L&15 from row `nrow` of a
// K-contiguous ("transposed") operand; k = 2v + 16*(L>>4) -> one 32B run.
__device__ __forceinline__ v16bf load_b_frag(const bf16* p, int ld, int nrow, int kk, int lane) {
  int hh = lane >> 4;
  const bf16* base = p + (long)nrow * ld + kk + 16 * hh;
  return *(const v16bf*)(base);
}
__device__ __forceinline__ v8f wmma_bf16(v16bf a, v16bf b, v8f c) {
  return __builtin_amdgcn_wmma_f32_16x16x32_bf16(false, a, false, b, (short)0, c, false, false);
}

// ---------------------------------------------------------------------------
// Kernel 1a: x fp32 -> bf16 (already coalesced).
// ---------------------------------------------------------------------------
__global__ void convert_x_kernel(const float* __restrict__ x, bf16* __restrict__ xb) {
  const int nx = M_ * C_;
  for (int i = blockIdx.x * blockDim.x + threadIdx.x; i < nx;
       i += gridDim.x * blockDim.x)
    xb[i] = f2bf(x[i]);
}

// ---------------------------------------------------------------------------
// Kernel 1b: weight transpose + fp32->bf16 via 32x32 LDS tiles.
// src [K][N] fp32 (coalesced reads) -> dst [N][K] bf16 (coalesced writes).
// ---------------------------------------------------------------------------
__global__ __launch_bounds__(256) void transpose_w_kernel(
    const float* __restrict__ src, bf16* __restrict__ dst, int K, int N) {
  __shared__ unsigned short tile[32][33];     // +1 pad vs bank conflicts
  int nT = blockIdx.x * 32;
  int kT = blockIdx.y * 32;
  int c  = threadIdx.x & 31;
  int r0 = threadIdx.x >> 5;                  // 0..7
  #pragma unroll
  for (int i = 0; i < 4; ++i) {
    int r = r0 + 8 * i;
    tile[r][c] = f2bf_u(src[(long)(kT + r) * N + nT + c]);
  }
  __syncthreads();
  #pragma unroll
  for (int i = 0; i < 4; ++i) {
    int r = r0 + 8 * i;                       // n within tile
    dst[(long)(nT + r) * K + kT + c] = __builtin_bit_cast(bf16, tile[c][r]);
  }
}

// ---------------------------------------------------------------------------
// Kernel 2: QKV GEMM. [4096,1024] x [1024,3072] + bias.
// 8 waves/block (4x2), each wave a 64x64 tile (4x4 WMMA per k-step).
// Outputs: Q (pre-scaled by 1/sqrt(dh)), K as [B,H,T,64]; V transposed [B,H,64,T].
// ---------------------------------------------------------------------------
__global__ __launch_bounds__(256) void gemm_qkv_kernel(
    const bf16* __restrict__ Xb, const bf16* __restrict__ WT,
    const float* __restrict__ bias,
    bf16* __restrict__ Qd, bf16* __restrict__ Kd, bf16* __restrict__ Vt) {
  int lane = threadIdx.x & 31;
  int wid  = threadIdx.x >> 5;
  int rowBase = blockIdx.x * 256 + (wid & 3) * 64;
  int colBase = blockIdx.y * 128 + (wid >> 2) * 64;
  int n = lane & 15, hh = lane >> 4;

  v8f acc[4][4] = {};
  #pragma unroll 2
  for (int kk = 0; kk < C_; kk += 32) {
    v16bf a[4];
    #pragma unroll
    for (int i = 0; i < 4; ++i)
      a[i] = load_a_frag(Xb, C_, rowBase + i * 16 + n, kk, lane);
    #pragma unroll
    for (int j = 0; j < 4; ++j) {
      v16bf b = load_b_frag(WT, C_, colBase + j * 16 + n, kk, lane);
      #pragma unroll
      for (int i = 0; i < 4; ++i) acc[i][j] = wmma_bf16(a[i], b, acc[i][j]);
    }
  }
  #pragma unroll
  for (int i = 0; i < 4; ++i)
  #pragma unroll
  for (int j = 0; j < 4; ++j)
  #pragma unroll
  for (int r = 0; r < 8; ++r) {
    int row = rowBase + i * 16 + r + 8 * hh;      // global token row (0..4095)
    int col = colBase + j * 16 + n;               // global feature (0..3071)
    float v = acc[i][j][r] + bias[col];
    int which = col >> 10, rem = col & (C_ - 1);
    int head = rem >> 6, d = rem & (DH_ - 1);
    int bb = row >> 11, t = row & (T_ - 1);
    long hb = (long)bb * H_ + head;
    if (which == 0)      Qd[(hb * T_ + t) * DH_ + d] = f2bf(v * 0.125f); // 1/sqrt(64)
    else if (which == 1) Kd[(hb * T_ + t) * DH_ + d] = f2bf(v);
    else                 Vt[(hb * DH_ + d) * T_ + t] = f2bf(v);
  }
}

// ---------------------------------------------------------------------------
// Kernel 3: causal attention, flash-style. One wave = 16 query rows.
// 128-key chunks: 16 WMMA (QK^T) + 16 WMMA (PV) per chunk; fp32 softmax.
// Next chunk's K/V prefetched (global_prefetch) to overlap with the scan.
// ---------------------------------------------------------------------------
__global__ __launch_bounds__(128) void attn_kernel(
    const bf16* __restrict__ Qd, const bf16* __restrict__ Kd,
    const bf16* __restrict__ Vt, bf16* __restrict__ Od) {
  __shared__ __align__(32) float          Sld[4][16][128];
  __shared__ __align__(32) unsigned short Pld[4][16][128];

  int lane = threadIdx.x & 31;
  int w    = threadIdx.x >> 5;
  int flat = blockIdx.x * 4 + w;           // 4096 query tiles total
  int qt = flat & 127;                     // T/16
  int h  = (flat >> 7) & (H_ - 1);
  int bb = flat >> 11;
  int n = lane & 15, hh = lane >> 4;

  const bf16* Qb = Qd + ((long)(bb * H_ + h) * T_) * DH_;
  const bf16* Kb = Kd + ((long)(bb * H_ + h) * T_) * DH_;
  const bf16* Vb = Vt + ((long)(bb * H_ + h) * DH_) * T_;

  // Q A-fragments for d = [0,32) and [32,64): held in VGPRs all kernel
  v16bf aQ0 = load_a_frag(Qb, DH_, qt * 16 + n, 0,  lane);
  v16bf aQ1 = load_a_frag(Qb, DH_, qt * 16 + n, 32, lane);

  v8f acc[4] = {};
  float mrow = -3.0e30f, lrow = 0.0f;      // stats for row (lane&15), dup'd per half
  int q_r = qt * 16 + n;
  int limit = (qt + 1) * 16;
  int nch = (limit + 127) >> 7;

  for (int ch = 0; ch < nch; ++ch) {
    int base = ch << 7;

    // --- prefetch next chunk's K rows and V columns into cache ---
    if (ch + 1 < nch) {
      int nb = base + 128;
      #pragma unroll
      for (int r = 0; r < 4; ++r)
        __builtin_prefetch(Kb + (long)(nb + lane * 4 + r) * DH_, 0, 1);
      #pragma unroll
      for (int s = 0; s < 2; ++s) {
        const bf16* vp = Vb + (long)(lane + 32 * s) * T_ + nb;
        __builtin_prefetch(vp, 0, 1);
        __builtin_prefetch(vp + 64, 0, 1);
      }
    }

    // --- scores: 8 key tiles of 16 ---
    for (int t8 = 0; t8 < 8; ++t8) {
      int kt = base + t8 * 16;
      v8f s = {};
      v16bf bK0 = load_b_frag(Kb, DH_, kt + n, 0,  lane);
      v16bf bK1 = load_b_frag(Kb, DH_, kt + n, 32, lane);
      s = wmma_bf16(aQ0, bK0, s);
      s = wmma_bf16(aQ1, bK1, s);
      #pragma unroll
      for (int r = 0; r < 8; ++r) Sld[w][r + 8 * hh][t8 * 16 + n] = s[r];
    }
    asm volatile("s_wait_dscnt 0" ::: "memory");   // in-wave LDS store->load order

    // --- online softmax scan: lane owns row (lane&15), cols [64*hh, 64*hh+64) ---
    float mx = -3.0e30f;
    #pragma unroll 4
    for (int c = 0; c < 64; ++c) {
      int col = hh * 64 + c;
      if (base + col <= q_r) mx = fmaxf(mx, Sld[w][n][col]);
    }
    mx = fmaxf(mx, lane_bcast(mx, lane ^ 16));
    float mnew  = fmaxf(mrow, mx);
    float scale = __expf(mrow - mnew);
    float sm = 0.0f;
    #pragma unroll 4
    for (int c = 0; c < 64; ++c) {
      int col = hh * 64 + c;
      float p = 0.0f;
      if (base + col <= q_r) { p = __expf(Sld[w][n][col] - mnew); sm += p; }
      Pld[w][n][col] = f2bf_u(p);
    }
    sm += lane_bcast(sm, lane ^ 16);
    lrow = lrow * scale + sm;
    mrow = mnew;
    asm volatile("s_wait_dscnt 0" ::: "memory");

    // --- rescale output accumulators by exp(m_old - m_new) per row ---
    #pragma unroll
    for (int r = 0; r < 8; ++r) {
      float sc = lane_bcast(scale, hh * 8 + r);
      #pragma unroll
      for (int dt = 0; dt < 4; ++dt) acc[dt][r] *= sc;
    }
    // --- P @ V : 4 k-steps x 4 d-tiles ---
    for (int ks = 0; ks < 4; ++ks) {
      int kk = ks * 32;
      v16bf aP = load_a_frag((const bf16*)&Pld[w][0][0], 128, n, kk, lane);
      #pragma unroll
      for (int dt = 0; dt < 4; ++dt) {
        v16bf bV = load_b_frag(Vb, T_, dt * 16 + n, base + kk, lane);
        acc[dt] = wmma_bf16(aP, bV, acc[dt]);
      }
    }
  }

  // --- normalize and store O as bf16 [B, T, C] (c = h*64 + dt*16 + n) ---
  #pragma unroll
  for (int r = 0; r < 8; ++r) {
    float inv = 1.0f / lane_bcast(lrow, hh * 8 + r);
    int qrow = qt * 16 + r + 8 * hh;
    #pragma unroll
    for (int dt = 0; dt < 4; ++dt)
      Od[((long)(bb * T_ + qrow)) * C_ + h * DH_ + dt * 16 + n] =
          f2bf(acc[dt][r] * inv);
  }
}

// ---------------------------------------------------------------------------
// Kernel 4: output projection. [4096,1024] x [1024,1024] + bias -> fp32 out.
// Same 64x64 wave tiling as the QKV GEMM.
// ---------------------------------------------------------------------------
__global__ __launch_bounds__(256) void gemm_proj_kernel(
    const bf16* __restrict__ Ob, const bf16* __restrict__ WT,
    const float* __restrict__ bias, float* __restrict__ out) {
  int lane = threadIdx.x & 31;
  int wid  = threadIdx.x >> 5;
  int rowBase = blockIdx.x * 256 + (wid & 3) * 64;
  int colBase = blockIdx.y * 128 + (wid >> 2) * 64;
  int n = lane & 15, hh = lane >> 4;

  v8f acc[4][4] = {};
  #pragma unroll 2
  for (int kk = 0; kk < C_; kk += 32) {
    v16bf a[4];
    #pragma unroll
    for (int i = 0; i < 4; ++i)
      a[i] = load_a_frag(Ob, C_, rowBase + i * 16 + n, kk, lane);
    #pragma unroll
    for (int j = 0; j < 4; ++j) {
      v16bf b = load_b_frag(WT, C_, colBase + j * 16 + n, kk, lane);
      #pragma unroll
      for (int i = 0; i < 4; ++i) acc[i][j] = wmma_bf16(a[i], b, acc[i][j]);
    }
  }
  #pragma unroll
  for (int i = 0; i < 4; ++i)
  #pragma unroll
  for (int j = 0; j < 4; ++j)
  #pragma unroll
  for (int r = 0; r < 8; ++r) {
    int row = rowBase + i * 16 + r + 8 * hh;
    int col = colBase + j * 16 + n;
    out[(long)row * C_ + col] = acc[i][j][r] + bias[col];
  }
}

// ---------------------------------------------------------------------------
extern "C" void kernel_launch(void* const* d_in, const int* in_sizes, int n_in,
                              void* d_out, int out_size, void* d_ws, size_t ws_size,
                              hipStream_t stream) {
  const float* x     = (const float*)d_in[0];   // [2,2048,1024]
  const float* wqkv  = (const float*)d_in[1];   // [1024,3072]
  const float* bqkv  = (const float*)d_in[2];   // [3072]
  const float* wproj = (const float*)d_in[3];   // [1024,1024]
  const float* bproj = (const float*)d_in[4];   // [1024]
  float* out = (float*)d_out;                   // [2,2048,1024] fp32

  // Workspace layout (~48 MB total, 256B aligned chunks)
  char* ws = (char*)d_ws;
  size_t off = 0;
  auto take = [&](size_t bytes) {
    char* p = ws + off;
    off += (bytes + 255) & ~(size_t)255;
    return p;
  };
  bf16* xb     = (bf16*)take((size_t)M_ * C_ * 2);           // 8 MB
  bf16* wqkvT  = (bf16*)take((size_t)N1_ * C_ * 2);          // 6 MB
  bf16* wprojT = (bf16*)take((size_t)C_ * C_ * 2);           // 2 MB
  bf16* Qd     = (bf16*)take((size_t)B_ * H_ * T_ * DH_ * 2);// 8 MB
  bf16* Kd     = (bf16*)take((size_t)B_ * H_ * T_ * DH_ * 2);// 8 MB
  bf16* Vtd    = (bf16*)take((size_t)B_ * H_ * T_ * DH_ * 2);// 8 MB
  bf16* Od     = (bf16*)take((size_t)M_ * C_ * 2);           // 8 MB
  (void)ws_size; (void)in_sizes; (void)n_in; (void)out_size;

  convert_x_kernel<<<2048, 256, 0, stream>>>(x, xb);
  transpose_w_kernel<<<dim3(N1_ / 32, C_ / 32), 256, 0, stream>>>(wqkv, wqkvT, C_, N1_);
  transpose_w_kernel<<<dim3(C_ / 32, C_ / 32), 256, 0, stream>>>(wproj, wprojT, C_, C_);
  gemm_qkv_kernel<<<dim3(M_ / 256, N1_ / 128), 256, 0, stream>>>(xb, wqkvT, bqkv, Qd, Kd, Vtd);
  attn_kernel<<<1024, 128, 0, stream>>>(Qd, Kd, Vtd, Od);
  gemm_proj_kernel<<<dim3(M_ / 256, C_ / 128), 256, 0, stream>>>(Od, wprojT, bproj, out);
}
